// DualAttention_83081847374417
// MI455X (gfx1250) — matrix operand order repaired and checked by reading, not compiled, not run on hardware
//
#include <hip/hip_runtime.h>
#include <hip/hip_bf16.h>
#include <math.h>

#define B_   32
#define N_   4096
#define DM   256
#define NH   8
#define DH   32
#define PROJ 40

typedef __attribute__((ext_vector_type(16))) _Float16 v16h;
typedef __attribute__((ext_vector_type(8)))  float    v8f;

// d_out layout (float offsets): logits, pooled, attn_out, query
#define LOGITS_OFF 0
#define POOLED_OFF (B_*PROJ)                 // 1280
#define ATTN_OFF   (POOLED_OFF + B_*DM)      // 9472
#define QUERY_OFF  (ATTN_OFF + B_*2*DM)      // 25856

// workspace byte offsets (total ~143.5 MB)
#define WS_Q   ((size_t)0)                               // 32*2*256*4      = 64 KB
#define WS_SC  (WS_Q  + (size_t)B_*2*DM*4)               // 32*8*2*4096*4   = 8 MB
#define WS_V   (WS_SC + (size_t)B_*NH*2*N_*4)            // 32*4096*256*4   = 128 MB
#define WS_W   (WS_V  + (size_t)B_*N_*DM*4)              // 3*2*65536*2     = 768 KB

#define AST  260     // LDS row stride (u32) for the 128x256 f32 tile
#define ROWW 72      // LDS row stride (u32) for hi/lo A planes (64 data + pad, 32B-mult)
#define TM   128     // rows per block tile
#define WMAT 131072  // halves per weight matrix (2 planes * 65536)

// split f32 -> (f16 hi, f16 lo); pack two adjacent elements per plane word
__device__ __forceinline__ void split2(float x0, float x1, unsigned& uh, unsigned& ul) {
  _Float16 h0 = (_Float16)x0, h1 = (_Float16)x1;
  _Float16 l0 = (_Float16)(x0 - (float)h0), l1 = (_Float16)(x1 - (float)h1);
  uh = ((unsigned)__builtin_bit_cast(unsigned short, h1) << 16) |
       (unsigned)__builtin_bit_cast(unsigned short, h0);
  ul = ((unsigned)__builtin_bit_cast(unsigned short, l1) << 16) |
       (unsigned)__builtin_bit_cast(unsigned short, l0);
}
// swizzle of pair-index within a 32K slab so each lane's A fragment is one
// contiguous 32B chunk: q 0-3 -> 0-3, 4-7 -> 8-11, 8-11 -> 4-7, 12-15 -> 12-15
__device__ __forceinline__ int pair_swz(int q) {
  return (q & 3) | ((q & 8) >> 1) | ((q & 4) << 1);
}

__device__ __forceinline__ v8f wmma_f16(v16h a, v16h b, v8f c) {
  return __builtin_amdgcn_wmma_f32_16x16x32_f16(false, a, false, b, (short)0, c,
                                                false, false);
}

// ---- half of a 128x256 GEMM: 8 col-tiles, acc[8] statically indexed ----
// A: hi/lo planes in LDS, fragment-swizzled -> one 32B read per plane.
// B: fragment-ordered halves in global (wf hi plane, wf+65536 lo plane).
__device__ __forceinline__ void gemm_half(const unsigned* __restrict__ aph,
                                          const unsigned* __restrict__ apl,
                                          const unsigned short* __restrict__ wf,
                                          v8f acc[8], int rowbase, int lane,
                                          int half) {
  const v16h* bhp = (const v16h*)wf;
  const v16h* blp = (const v16h*)(wf + 65536);
  int row = rowbase + (lane & 15);
  int aoff = row * ROWW + ((lane >> 4) << 3);
#pragma unroll 1
  for (int kk8 = 0; kk8 < 8; ++kk8) {
    v16h ahi = *(const v16h*)(aph + aoff + (kk8 << 4));
    v16h alo = *(const v16h*)(apl + aoff + (kk8 << 4));
    const v16h* bh = bhp + ((kk8 * 16 + half * 8) * 32 + lane);
    const v16h* bl = blp + ((kk8 * 16 + half * 8) * 32 + lane);
#pragma unroll
    for (int jj = 0; jj < 8; ++jj) {
      v16h bhi = bh[jj * 32];
      v16h blo = bl[jj * 32];
      // fp32-ish product: hi*hi + hi*lo + lo*hi
      acc[jj] = wmma_f16(ahi, bhi, acc[jj]);
      acc[jj] = wmma_f16(ahi, blo, acc[jj]);
      acc[jj] = wmma_f16(alo, bhi, acc[jj]);
    }
  }
}

// full 128x256 GEMM (+bias) from hi/lo planes into dst (LDS, f32 bits)
__device__ __forceinline__ void gemm_dump(const unsigned* __restrict__ aph,
                                          const unsigned* __restrict__ apl,
                                          const unsigned short* __restrict__ wf,
                                          unsigned* __restrict__ dst,
                                          const float* __restrict__ bias,
                                          int rowbase, int lane) {
  int colb = lane & 15, hrow = (lane >> 4) << 3;
#pragma unroll 1
  for (int half = 0; half < 2; ++half) {
    v8f acc[8];
#pragma unroll
    for (int j = 0; j < 8; ++j) {
      v8f z = {0, 0, 0, 0, 0, 0, 0, 0};
      acc[j] = z;
    }
    gemm_half(aph, apl, wf, acc, rowbase, lane, half);
#pragma unroll
    for (int jj = 0; jj < 8; ++jj) {
      int col = ((half * 8 + jj) << 4) + colb;
      float bb = bias[col];
#pragma unroll
      for (int r = 0; r < 8; ++r)
        dst[(rowbase + hrow + r) * AST + col] = __float_as_uint(acc[jj][r] + bb);
    }
  }
}

// ---------------- K0: pack weights into fragment-ordered hi/lo planes --------
// layout: wfh[m][plane][kk8][j][lane][i] halves; element (kk8,j,lane,i) maps to
// W[k][n] with n = 16j + lane%16, k = 32*kk8 + 8*(lane>=16) + i%8 + 16*(i>=8)
__global__ void k0_pack(const float* __restrict__ eeg_w,
                        const float* __restrict__ k_w,
                        const float* __restrict__ v_w,
                        unsigned short* __restrict__ wfh) {
  int idx = blockIdx.x * 256 + threadIdx.x;   // 3*65536 total
  int m = idx >> 16, e = idx & 0xffff;
  int i = e & 15, lane = (e >> 4) & 31, j = (e >> 9) & 15, kk8 = e >> 13;
  int n = (j << 4) + (lane & 15);
  int k = (kk8 << 5) + ((lane >> 4) << 3) + (i & 7) + ((i >> 3) << 4);
  const float* s = (m == 0) ? eeg_w : (m == 1) ? k_w : v_w;
  float x = s[k * DM + n];
  _Float16 h = (_Float16)x;
  _Float16 l = (_Float16)(x - (float)h);
  wfh[(size_t)m * WMAT + e]         = __builtin_bit_cast(unsigned short, h);
  wfh[(size_t)m * WMAT + 65536 + e] = __builtin_bit_cast(unsigned short, l);
}

// ---------------- K1: query path (tiny) ----------------
__global__ __launch_bounds__(256)
void k1_query(const float* __restrict__ stim_feat, const float* __restrict__ temp_feat,
              const float* __restrict__ stim_w, const float* __restrict__ stim_b,
              const float* __restrict__ temp_w, const float* __restrict__ temp_b,
              const float* __restrict__ q_w, const float* __restrict__ q_b,
              const float* __restrict__ lnq_g, const float* __restrict__ lnq_b,
              float* __restrict__ wsQ, float* __restrict__ out) {
  __shared__ float sb[2][DM];
  __shared__ float qln[2][DM];
  __shared__ float qp[2][DM];
  __shared__ float mv[4];
  int b = blockIdx.x, c = threadIdx.x;
  float a0 = stim_b[c], a1 = temp_b[c];
  for (int k = 0; k < DM; ++k) {
    a0 += stim_feat[b * DM + k] * stim_w[k * DM + c];
    a1 += temp_feat[b * DM + k] * temp_w[k * DM + c];
  }
  sb[0][c] = a0; sb[1][c] = a1;
  __syncthreads();
  if (c < 2) {
    float s = 0, ss = 0;
    for (int k = 0; k < DM; ++k) { float x = sb[c][k]; s += x; ss += x * x; }
    float m = s * (1.f / DM);
    mv[c] = m; mv[2 + c] = ss * (1.f / DM) - m * m;
  }
  __syncthreads();
  for (int q = 0; q < 2; ++q) {
    float y = (sb[q][c] - mv[q]) * rsqrtf(mv[2 + q] + 1e-5f) * lnq_g[c] + lnq_b[c];
    qln[q][c] = y;
    out[QUERY_OFF + (b * 2 + q) * DM + c] = y;   // "query" output
  }
  __syncthreads();
  for (int q = 0; q < 2; ++q) {
    float v = q_b[c];
    for (int k = 0; k < DM; ++k) v += qln[q][k] * q_w[k * DM + c];
    qp[q][c] = v;
  }
  __syncthreads();
  if (c < 16) {  // rope(pos=q) + l2norm per (q,h)
    int q = c >> 3, h = c & 7;
    float x[DH];
#pragma unroll
    for (int d = 0; d < DH; ++d) x[d] = qp[q][h * DH + d];
#pragma unroll
    for (int i = 0; i < 16; ++i) {
      float inv = exp2f(-(float)i * 0.8304820237218405f);  // log2(1e4)/16
      float ang = (float)q * inv;
      float cc = cosf(ang), sn = sinf(ang);
      float x0 = x[2 * i], x1 = x[2 * i + 1];
      x[2 * i] = x0 * cc - x1 * sn;
      x[2 * i + 1] = x1 * cc + x0 * sn;
    }
    float nr = 0;
#pragma unroll
    for (int d = 0; d < DH; ++d) nr += x[d] * x[d];
    float r = 1.f / fmaxf(sqrtf(nr), 1e-12f);
#pragma unroll
    for (int d = 0; d < DH; ++d) wsQ[(b * 2 + q) * DM + h * DH + d] = x[d] * r;
  }
}

// ---------------- K2: fused X@We -> LN -> K,V GEMMs + rope/scores + V dump ----
__global__ __launch_bounds__(256)
void k2_main(const float* __restrict__ eeg_feat,
             const float* __restrict__ eeg_b,
             const float* __restrict__ k_b,
             const float* __restrict__ v_b,
             const float* __restrict__ lnk_g,
             const float* __restrict__ lnk_b,
             const unsigned short* __restrict__ wfrag,
             const float* __restrict__ wsQ,
             float* __restrict__ wsSC,
             float* __restrict__ wsV) {
  // f32 results tile (C1 -> K -> V)
  __shared__ __attribute__((aligned(32))) unsigned fbuf[TM * AST];
  // hi/lo f16 A-operand planes (X, then kv), fragment-swizzled
  __shared__ __attribute__((aligned(32))) unsigned planes[2 * TM * ROWW];
  __shared__ float qbuf[2 * DM];

  unsigned* aph = planes;
  unsigned* apl = planes + TM * ROWW;

  int tid = threadIdx.x, lane = tid & 31, wave = tid >> 5;
  int b = blockIdx.x >> 5, tile = blockIdx.x & 31;
  int rowbase = wave << 4;
  int n0 = tile * TM;

  qbuf[tid] = wsQ[b * 2 * DM + tid];
  qbuf[tid + 256] = wsQ[b * 2 * DM + 256 + tid];

  // ---- load X tile, split hi/lo, store swizzled planes (coalesced float2)
  const float2* xp2 = (const float2*)(eeg_feat + ((size_t)b * N_ + n0) * DM);
  for (int i = 0; i < 64; ++i) {
    int f = tid + (i << 8);                 // 16384 pairs total
    int row = f >> 7, pc = f & 127;
    int kk8 = pc >> 4, q = pc & 15;
    int dst = row * ROWW + (kk8 << 4) + pair_swz(q);
    float2 x = xp2[(size_t)row * (DM / 2) + pc];
    unsigned uh, ul; split2(x.x, x.y, uh, ul);
    aph[dst] = uh; apl[dst] = ul;
  }
  __syncthreads();

  // ---- GEMM1: C1 = X @ eeg_w (+bias) -> fbuf (f32 bits)
  gemm_dump(aph, apl, wfrag + 0 * (size_t)WMAT, fbuf, eeg_b, rowbase, lane);
  __syncthreads();

  // ---- LayerNorm rows: fbuf f32 -> kv planes (overwrite X planes)
  if (tid < TM) {
    int row = tid;
    float s = 0, ss = 0;
    for (int c = 0; c < DM; ++c) {
      float x = __uint_as_float(fbuf[row * AST + c]);
      s += x; ss += x * x;
    }
    float mean = s * (1.f / DM);
    float var = ss * (1.f / DM) - mean * mean;
    float rstd = rsqrtf(var + 1e-5f);
    for (int pc = 0; pc < 128; ++pc) {
      float x0 = __uint_as_float(fbuf[row * AST + 2 * pc]);
      float x1 = __uint_as_float(fbuf[row * AST + 2 * pc + 1]);
      float y0 = (x0 - mean) * rstd * lnk_g[2 * pc] + lnk_b[2 * pc];
      float y1 = (x1 - mean) * rstd * lnk_g[2 * pc + 1] + lnk_b[2 * pc + 1];
      int dst = row * ROWW + ((pc >> 4) << 4) + pair_swz(pc & 15);
      unsigned uh, ul; split2(y0, y1, uh, ul);
      aph[dst] = uh; apl[dst] = ul;
    }
  }
  __syncthreads();

  // ---- GEMM2: K = kv @ k_w (+bias) -> fbuf (f32 bits)
  gemm_dump(aph, apl, wfrag + 1 * (size_t)WMAT, fbuf, k_b, rowbase, lane);
  __syncthreads();

  // ---- RoPE + l2norm + scores = Qn . Kn / sqrt(32)
  for (int it = 0; it < 4; ++it) {
    int p = tid + (it << 8);
    int h = p >> 7, row = p & 127;
    int n = n0 + row;
    float x[DH];
#pragma unroll
    for (int d = 0; d < DH; ++d)
      x[d] = __uint_as_float(fbuf[row * AST + h * DH + d]);
#pragma unroll
    for (int i = 0; i < 16; ++i) {
      float inv = exp2f(-(float)i * 0.8304820237218405f);
      float ang = (float)n * inv;
      float cc = cosf(ang), sn = sinf(ang);
      float x0 = x[2 * i], x1 = x[2 * i + 1];
      x[2 * i] = x0 * cc - x1 * sn;
      x[2 * i + 1] = x1 * cc + x0 * sn;
    }
    float nr = 0;
#pragma unroll
    for (int d = 0; d < DH; ++d) nr += x[d] * x[d];
    float rinv = 1.f / fmaxf(sqrtf(nr), 1e-12f);
#pragma unroll
    for (int q = 0; q < 2; ++q) {
      float dot = 0;
#pragma unroll
      for (int d = 0; d < DH; ++d) dot += x[d] * qbuf[q * DM + h * DH + d];
      wsSC[((size_t)((b * NH + h) * 2 + q)) * N_ + n] =
          dot * rinv * 0.17677669529663687f;  // 1/sqrt(32)
    }
  }
  __syncthreads();

  // ---- GEMM3: V = kv @ v_w (+bias) -> fbuf -> global (coalesced)
  gemm_dump(aph, apl, wfrag + 2 * (size_t)WMAT, fbuf, v_b, rowbase, lane);
  __syncthreads();
  float* vout = wsV + ((size_t)b * N_ + n0) * DM;
  for (int r = 0; r < TM; ++r)
    vout[(size_t)r * DM + tid] = __uint_as_float(fbuf[r * AST + tid]);
}

// ---------------- K3: std / tanh-clip / softmax / renorm (in-place) ----------
__global__ __launch_bounds__(256)
void k3_softmax(float* __restrict__ wsSC) {
  __shared__ float red[256];
  int bh = blockIdx.x, tid = threadIdx.x;
  for (int q = 0; q < 2; ++q) {
    float* s = wsSC + ((size_t)(bh * 2 + q)) * N_;
    float lv[16];
    float sm = 0, ss = 0;
#pragma unroll
    for (int i = 0; i < 16; ++i) {
      float x = s[tid + (i << 8)];
      lv[i] = x; sm += x; ss += x * x;
    }
    red[tid] = sm; __syncthreads();
    for (int o = 128; o > 0; o >>= 1) { if (tid < o) red[tid] += red[tid + o]; __syncthreads(); }
    sm = red[0]; __syncthreads();
    red[tid] = ss; __syncthreads();
    for (int o = 128; o > 0; o >>= 1) { if (tid < o) red[tid] += red[tid + o]; __syncthreads(); }
    ss = red[0]; __syncthreads();

    float mean = sm * (1.f / N_);
    float var = (ss - (float)N_ * mean * mean) * (1.f / (N_ - 1));  // ddof=1
    float sig = fmaxf(sqrtf(fmaxf(var, 0.f)), 1e-3f);
    float i2s = 0.5f / sig, amp = 5.f * sig;

    float mx = -3.4e38f;
#pragma unroll
    for (int i = 0; i < 16; ++i) { lv[i] = tanhf(lv[i] * i2s) * amp; mx = fmaxf(mx, lv[i]); }
    red[tid] = mx; __syncthreads();
    for (int o = 128; o > 0; o >>= 1) { if (tid < o) red[tid] = fmaxf(red[tid], red[tid + o]); __syncthreads(); }
    mx = red[0]; __syncthreads();

    float es = 0;
#pragma unroll
    for (int i = 0; i < 16; ++i) { lv[i] = __expf(lv[i] - mx); es += lv[i]; }
    red[tid] = es; __syncthreads();
    for (int o = 128; o > 0; o >>= 1) { if (tid < o) red[tid] += red[tid + o]; __syncthreads(); }
    es = red[0]; __syncthreads();

    float invS = 1.f / es, cn = 1.f / (1.f + 1e-8f);
#pragma unroll
    for (int i = 0; i < 16; ++i)
      s[tid + (i << 8)] = (lv[i] * invS + 1e-4f) * cn;  // (w+1e-4)/(1+1e-8)
    __syncthreads();
  }
}

// ---------------- K4: attn_out = w @ V, pooled, logits ----------------
__global__ __launch_bounds__(256)
void k4_attn(const float* __restrict__ wsSC, const float* __restrict__ wsV,
             const float* __restrict__ fc_w, const float* __restrict__ fc_b,
             float* __restrict__ out) {
  __shared__ float wt[16 * 128];   // [q][h][kk]
  __shared__ float pooled[DM];
  int b = blockIdx.x, tid = threadIdx.x;
  int h = tid >> 5;
  float acc0 = 0, acc1 = 0;
  for (int kt = 0; kt < N_ / 128; ++kt) {
    __syncthreads();
#pragma unroll
    for (int i = 0; i < 8; ++i) {
      int f = tid + (i << 8);
      int kk = f & 127, hq = f >> 7;
      int hh = hq >> 1, q = hq & 1;
      wt[(q * 8 + hh) * 128 + kk] =
          wsSC[((size_t)((b * NH + hh) * 2 + q)) * N_ + kt * 128 + kk];
    }
    __syncthreads();
    const float* vp = wsV + ((size_t)b * N_ + kt * 128) * DM + tid;
    const float* w0 = &wt[h * 128];
    const float* w1 = &wt[(8 + h) * 128];
    for (int kk = 0; kk < 128; ++kk) {
      float v = vp[(size_t)kk * DM];
      acc0 += w0[kk] * v;
      acc1 += w1[kk] * v;
    }
  }
  out[ATTN_OFF + (b * 2 + 0) * DM + tid] = acc0;
  out[ATTN_OFF + (b * 2 + 1) * DM + tid] = acc1;
  float p = 0.5f * (acc0 + acc1);
  pooled[tid] = p;
  out[POOLED_OFF + b * DM + tid] = p;
  __syncthreads();
  if (tid < PROJ) {
    float lg = fc_b[tid];
    for (int c = 0; c < DM; ++c) lg += pooled[c] * fc_w[c * PROJ + tid];
    out[LOGITS_OFF + b * PROJ + tid] = lg;
  }
}

extern "C" void kernel_launch(void* const* d_in, const int* in_sizes, int n_in,
                              void* d_out, int out_size, void* d_ws, size_t ws_size,
                              hipStream_t stream) {
  (void)in_sizes; (void)n_in; (void)out_size; (void)ws_size;
  const float* eeg_feat  = (const float*)d_in[0];
  const float* stim_feat = (const float*)d_in[1];
  const float* temp_feat = (const float*)d_in[2];
  const float* eeg_w     = (const float*)d_in[3];
  const float* eeg_b     = (const float*)d_in[4];
  const float* stim_w    = (const float*)d_in[5];
  const float* stim_b    = (const float*)d_in[6];
  const float* temp_w    = (const float*)d_in[7];
  const float* temp_b    = (const float*)d_in[8];
  const float* q_w       = (const float*)d_in[9];
  const float* q_b       = (const float*)d_in[10];
  const float* k_w       = (const float*)d_in[11];
  const float* k_b       = (const float*)d_in[12];
  const float* v_w       = (const float*)d_in[13];
  const float* v_b       = (const float*)d_in[14];
  const float* fc_w      = (const float*)d_in[15];
  const float* fc_b      = (const float*)d_in[16];
  const float* lnq_g     = (const float*)d_in[17];
  const float* lnq_b     = (const float*)d_in[18];
  const float* lnk_g     = (const float*)d_in[19];
  const float* lnk_b     = (const float*)d_in[20];

  char* ws = (char*)d_ws;
  float*          wsQ   = (float*)(ws + WS_Q);
  float*          wsSC  = (float*)(ws + WS_SC);
  float*          wsV   = (float*)(ws + WS_V);
  unsigned short* wfrag = (unsigned short*)(ws + WS_W);
  float* out = (float*)d_out;

  k0_pack<<<768, 256, 0, stream>>>(eeg_w, k_w, v_w, wfrag);
  k1_query<<<B_, 256, 0, stream>>>(stim_feat, temp_feat, stim_w, stim_b,
                                   temp_w, temp_b, q_w, q_b, lnq_g, lnq_b,
                                   wsQ, out);
  k2_main<<<B_ * (N_ / TM), 256, 0, stream>>>(eeg_feat, eeg_b, k_b, v_b,
                                              lnk_g, lnk_b, wfrag, wsQ,
                                              wsSC, wsV);
  k3_softmax<<<B_ * NH, 256, 0, stream>>>(wsSC);
  k4_attn<<<B_, 256, 0, stream>>>(wsSC, wsV, fc_w, fc_b, out);
}
